// GraphSAINTGAT_74208444940495
// MI455X (gfx1250) — matrix operand order repaired and checked by reading, not compiled, not run on hardware
//
#include <hip/hip_runtime.h>
#include <hip/hip_bf16.h>
#include <math.h>

// ---------------------------------------------------------------------------
// GraphSAINT-style 3-layer GAT for MI455X (gfx1250, wave32).
//   layer0: 128 -> 32 x 4 heads (concat) + ELU
//   layer1: 128 -> 32 x 4 heads (concat) + ELU
//   layer2: 128 -> 40, 1 head (mean==identity) + log_softmax
// GEMMs run on v_wmma_f32_16x16x32_bf16 (bf16 in, f32 acc); edge softmax and
// aggregation stay fp32 (atomics land in the 192MB L2).
// ---------------------------------------------------------------------------

typedef __attribute__((ext_vector_type(16))) __bf16 v16bf;
typedef __attribute__((ext_vector_type(8)))  float  v8f;

#define NEG_SLOPE 0.2f

static __device__ __forceinline__ unsigned short f2bf(float f) {
  // round-to-nearest-even fp32 -> bf16 (NaN subtleties ignored for finite data)
  unsigned int u = __float_as_uint(f);
  u += 0x7FFFu + ((u >> 16) & 1u);
  return (unsigned short)(u >> 16);
}

union BF16V { v16bf v; unsigned short u[16]; };

static __device__ __forceinline__ void atomicMaxF(float* addr, float val) {
  // monotonic-int trick; buffer initialized to -inf
  if (val >= 0.0f)
    atomicMax((int*)addr, __float_as_int(val));
  else
    atomicMin((unsigned int*)addr, __float_as_uint(val));
}

// ---------------------------------------------------------------------------
// H[N,M] = X[N,128] * W[M,128]^T   (bf16 WMMA, f32 accumulate)
// 8 waves per block; wave w computes columns [16w,16w+16) of a 16-row tile.
// ---------------------------------------------------------------------------
__global__ __launch_bounds__(256) void gemm_wmma_bf16(
    const float* __restrict__ X, const float* __restrict__ W,
    float* __restrict__ Hout, int N, int M) {
  constexpr int K = 128;
  __shared__ unsigned short lA[16 * K];       // 4 KB
  __shared__ unsigned short lB[8][16 * K];    // 32 KB
  const int t       = threadIdx.x;
  const int wv      = t >> 5;
  const int lane    = t & 31;
  const int rowBase = blockIdx.x * 16;
  const int Mtiles  = (M + 15) >> 4;

  // Stage A tile (16 x 128) fp32 -> bf16, coalesced
  for (int idx = t; idx < 16 * K; idx += 256) {
    int r = idx >> 7, k = idx & (K - 1);
    int gr = rowBase + r;
    lA[idx] = f2bf(gr < N ? X[(size_t)gr * K + k] : 0.0f);
  }
  // Stage this wave's B tile (16 cols x 128 K) = rows of W
  if (wv < Mtiles) {
    const int colBase = wv << 4;
    for (int idx = lane; idx < 16 * K; idx += 32) {
      int r = idx >> 7, k = idx & (K - 1);
      int gm = colBase + r;
      lB[wv][idx] = f2bf(gm < M ? W[(size_t)gm * K + k] : 0.0f);
    }
  }
  __syncthreads();

  if (wv < Mtiles) {
    v8f acc = {0.f, 0.f, 0.f, 0.f, 0.f, 0.f, 0.f, 0.f};
    const int half = lane >> 4;   // which 16-lane half of the wave
    const int arow = lane & 15;   // A row (output row within tile)
    const int bcol = lane & 15;   // B column (output col within tile)
#pragma unroll
    for (int kk = 0; kk < K; kk += 32) {
      BF16V A, B;
#pragma unroll
      for (int i = 0; i < 16; ++i) {
        // ISA 16-bit A 16x32 layout: lanes<16 hold K {0..7,16..23}, lanes>=16 {8..15,24..31}
        const int ka = ((i >> 3) << 4) + (half << 3) + (i & 7);
        A.u[i] = lA[arow * K + kk + ka];
        // B 32x16: same K order as A: lanes<16 hold K {0..7,16..23}, lanes>=16 {8..15,24..31}
        const int kb = ((i >> 3) << 4) + (half << 3) + (i & 7);
        B.u[i] = lB[wv][bcol * K + kk + kb];
      }
      acc = __builtin_amdgcn_wmma_f32_16x16x32_bf16(
          false, A.v, false, B.v, (short)0, acc, false, false);
    }
    // C/D layout: VGPR r -> row r (lanes 0-15) / row 8+r (lanes 16-31), col = lane&15
    const int gcol = (wv << 4) + bcol;
#pragma unroll
    for (int r = 0; r < 8; ++r) {
      const int grow = rowBase + r + half * 8;
      if (grow < N && gcol < M) Hout[(size_t)grow * M + gcol] = acc[r];
    }
  }
}

// ---------------------------------------------------------------------------
// al_s[n,h] = sum_c h[n,h,c]*a_src[h,c] ; al_d likewise
// ---------------------------------------------------------------------------
__global__ void attn_scores(const float* __restrict__ hbuf,
                            const float* __restrict__ a_src,
                            const float* __restrict__ a_dst,
                            float* __restrict__ als, float* __restrict__ ald,
                            int N, int H, int C) {
  int t = blockIdx.x * blockDim.x + threadIdx.x;
  if (t >= N * H) return;
  int n = t / H, h = t - n * H;
  const float* hp = hbuf + (size_t)n * H * C + (size_t)h * C;
  const float* as = a_src + (size_t)h * C;
  const float* ad = a_dst + (size_t)h * C;
  float s = 0.f, d = 0.f;
  for (int c = 0; c < C; ++c) {
    float hv = hp[c];
    s += hv * as[c];
    d += hv * ad[c];
  }
  als[t] = s;
  ald[t] = d;
}

__global__ void init_bufs(float* __restrict__ agg, float* __restrict__ m,
                          float* __restrict__ den, int NHC, int NH) {
  int t = blockIdx.x * blockDim.x + threadIdx.x;
  if (t < NHC) agg[t] = 0.0f;
  if (t < NH) { m[t] = -__builtin_inff(); den[t] = 0.0f; }
}

// pass 1: e = leaky_relu(al_s[src]+al_d[dst]); segment max over dst
__global__ void edge_score_max(const int* __restrict__ ei, int Ein, int Etot,
                               const float* __restrict__ als,
                               const float* __restrict__ ald,
                               float* __restrict__ ebuf, float* __restrict__ m,
                               int H) {
  int t = blockIdx.x * blockDim.x + threadIdx.x;
  if (t >= Etot * H) return;
  int e = t / H, h = t - e * H;
  int s = (e < Ein) ? ei[e] : (e - Ein);
  int d = (e < Ein) ? ei[Ein + e] : (e - Ein);
  float v = als[s * H + h] + ald[d * H + h];
  v = v > 0.f ? v : NEG_SLOPE * v;
  ebuf[t] = v;
  atomicMaxF(&m[d * H + h], v);
}

// pass 2: ex = exp(e - m[dst]); segment sum over dst
__global__ void edge_exp_sum(const int* __restrict__ ei, int Ein, int Etot,
                             float* __restrict__ ebuf,
                             const float* __restrict__ m,
                             float* __restrict__ den, int H) {
  int t = blockIdx.x * blockDim.x + threadIdx.x;
  if (t >= Etot * H) return;
  int e = t / H, h = t - e * H;
  int d = (e < Ein) ? ei[Ein + e] : (e - Ein);
  float xv = __expf(ebuf[t] - m[d * H + h]);
  ebuf[t] = xv;
  atomicAdd(&den[d * H + h], xv);
}

// pass 3: agg[dst,h,c] += ex[e,h] * h[src,h,c]   (normalize later)
__global__ void edge_aggregate(const int* __restrict__ ei, int Ein, int Etot,
                               const float* __restrict__ ex,
                               const float* __restrict__ hfeat,
                               float* __restrict__ agg, int H, int C) {
  int e = blockIdx.x;
  if (e >= Etot) return;
  int s = (e < Ein) ? ei[e] : (e - Ein);
  int d = (e < Ein) ? ei[Ein + e] : (e - Ein);
  int HC = H * C;
  for (int c = threadIdx.x; c < HC; c += blockDim.x) {
    int h = c / C;
    float a = ex[e * H + h];
    atomicAdd(&agg[(size_t)d * HC + c], a * hfeat[(size_t)s * HC + c]);
  }
}

// layers 0/1: out = elu(agg/den + bias)
__global__ void finalize_elu(const float* __restrict__ agg,
                             const float* __restrict__ den,
                             const float* __restrict__ b,
                             float* __restrict__ feat, int N, int H, int C) {
  int HC = H * C;
  int t = blockIdx.x * blockDim.x + threadIdx.x;
  if (t >= N * HC) return;
  int n = t / HC, c = t - n * HC, h = c / C;
  float v = agg[t] / den[n * H + h] + b[c];
  feat[t] = v > 0.f ? v : (__expf(v) - 1.0f);
}

// layer 2: logits = agg/den + bias, then row-wise log_softmax (C=40, 64 thr/node)
__global__ __launch_bounds__(64) void finalize_logsoftmax(
    const float* __restrict__ agg, const float* __restrict__ den,
    const float* __restrict__ b, float* __restrict__ out, int N, int C) {
  __shared__ float sh[64];
  int n = blockIdx.x;
  int c = threadIdx.x;
  float val = 0.0f;
  float key = -__builtin_inff();
  if (c < C) {
    val = agg[(size_t)n * C + c] / den[n] + b[c];
    key = val;
  }
  sh[c] = key;
  __syncthreads();
  for (int s = 32; s > 0; s >>= 1) {
    if (c < s) sh[c] = fmaxf(sh[c], sh[c + s]);
    __syncthreads();
  }
  float mx = sh[0];
  __syncthreads();
  float ev = (c < C) ? __expf(val - mx) : 0.0f;
  sh[c] = ev;
  __syncthreads();
  for (int s = 32; s > 0; s >>= 1) {
    if (c < s) sh[c] += sh[c + s];
    __syncthreads();
  }
  float lse = __logf(sh[0]);
  if (c < C) out[(size_t)n * C + c] = (val - mx) - lse;
}

// ---------------------------------------------------------------------------
extern "C" void kernel_launch(void* const* d_in, const int* in_sizes, int n_in,
                              void* d_out, int out_size, void* d_ws, size_t ws_size,
                              hipStream_t stream) {
  const float* x   = (const float*)d_in[0];
  const int*   ei  = (const int*)d_in[1];   // [2,E] src row then dst row
  const float* W0  = (const float*)d_in[2];
  const float* as0 = (const float*)d_in[3];
  const float* ad0 = (const float*)d_in[4];
  const float* b0  = (const float*)d_in[5];
  const float* W1  = (const float*)d_in[6];
  const float* as1 = (const float*)d_in[7];
  const float* ad1 = (const float*)d_in[8];
  const float* b1  = (const float*)d_in[9];
  const float* W2  = (const float*)d_in[10];
  const float* as2 = (const float*)d_in[11];
  const float* ad2 = (const float*)d_in[12];
  const float* b2  = (const float*)d_in[13];

  const int N    = in_sizes[0] / 128;   // 50000
  const int Ein  = in_sizes[1] / 2;     // 800000
  const int Etot = Ein + N;             // + self loops

  // Carve fp32 workspace (~94 MB), 256B-aligned chunks
  float* base = (float*)d_ws;
  size_t off = 0;
  auto carve = [&](size_t n) {
    float* r = base + off;
    off += (n + 63) & ~(size_t)63;
    return r;
  };
  float* hbuf = carve((size_t)N * 128);    // post-GEMM features
  float* agg  = carve((size_t)N * 128);    // unnormalized aggregate
  float* feat = carve((size_t)N * 128);    // ELU output / next-layer input
  float* ebuf = carve((size_t)Etot * 4);   // edge score -> exp numerator
  float* als  = carve((size_t)N * 4);
  float* ald  = carve((size_t)N * 4);
  float* mbuf = carve((size_t)N * 4);
  float* den  = carve((size_t)N * 4);
  (void)ws_size; (void)n_in; (void)out_size;

  auto run_layer = [&](const float* in, const float* Wm, const float* asv,
                       const float* adv, const float* bv, int H, int C,
                       float* outFeat, float* logits) {
    const int HC  = H * C;
    const int nh  = N * H;
    const int nhc = N * HC;
    const int eh  = Etot * H;
    gemm_wmma_bf16<<<dim3((N + 15) / 16), dim3(256), 0, stream>>>(in, Wm, hbuf, N, HC);
    attn_scores<<<dim3((nh + 255) / 256), dim3(256), 0, stream>>>(hbuf, asv, adv, als, ald, N, H, C);
    init_bufs<<<dim3((nhc + 255) / 256), dim3(256), 0, stream>>>(agg, mbuf, den, nhc, nh);
    edge_score_max<<<dim3((eh + 255) / 256), dim3(256), 0, stream>>>(ei, Ein, Etot, als, ald, ebuf, mbuf, H);
    edge_exp_sum<<<dim3((eh + 255) / 256), dim3(256), 0, stream>>>(ei, Ein, Etot, ebuf, mbuf, den, H);
    edge_aggregate<<<dim3(Etot), dim3(HC >= 128 ? 128 : 64), 0, stream>>>(ei, Ein, Etot, ebuf, hbuf, agg, H, C);
    if (outFeat)
      finalize_elu<<<dim3((nhc + 255) / 256), dim3(256), 0, stream>>>(agg, den, bv, outFeat, N, H, C);
    else
      finalize_logsoftmax<<<dim3(N), dim3(64), 0, stream>>>(agg, den, bv, logits, N, C);
  };

  run_layer(x,    W0, as0, ad0, b0, 4, 32, feat, nullptr);
  run_layer(feat, W1, as1, ad1, b1, 4, 32, feat, nullptr);
  run_layer(feat, W2, as2, ad2, b2, 1, 40, nullptr, (float*)d_out);
}
